// MultiheadAttention_6725918785959
// MI455X (gfx1250) — matrix operand order repaired and checked by reading, not compiled, not run on hardware
//
#include <hip/hip_runtime.h>
#include <hip/hip_bf16.h>
#include <stdint.h>

#define S_LEN   2048
#define D_MODEL 768
#define NHEADS  12
#define DHEAD   64

typedef __attribute__((ext_vector_type(16))) __bf16 v16bf;
typedef __attribute__((ext_vector_type(8)))  __bf16 bf16x8;
typedef __attribute__((ext_vector_type(4)))  __bf16 bf16x4;
typedef __attribute__((ext_vector_type(8)))  float  v8f;

union Frag16 { v16bf v; bf16x8 h[2]; bf16x4 q[4]; };

__device__ __forceinline__ __bf16 f2bf(float f) {
  union { float f; uint32_t u; } v; v.f = f;
  uint32_t r = v.u + 0x7FFFu + ((v.u >> 16) & 1u);
  uint16_t hh = (uint16_t)(r >> 16);
  __bf16 o;
  __builtin_memcpy(&o, &hh, 2);
  return o;
}

__device__ __forceinline__ uint32_t lds_addr(const void* p) {
  // LDS aperture: flat addr[31:0] is the LDS byte address (ISA 10.2)
  return (uint32_t)(uintptr_t)p;
}

// ---------------------------------------------------------------------------
// fp32 -> bf16 conversion
// ---------------------------------------------------------------------------
__global__ __launch_bounds__(256)
void cvt_bf16_kernel(const float* __restrict__ src, __bf16* __restrict__ dst, int n) {
  int i = (blockIdx.x * blockDim.x + threadIdx.x) * 4;
  if (i + 3 < n) {
    float4 f = *(const float4*)(src + i);
    dst[i + 0] = f2bf(f.x);
    dst[i + 1] = f2bf(f.y);
    dst[i + 2] = f2bf(f.z);
    dst[i + 3] = f2bf(f.w);
  } else {
    for (; i < n; ++i) dst[i] = f2bf(src[i]);
  }
}

// ---------------------------------------------------------------------------
// wave-level GEMM: one wave computes a 16(M) x 64(N) fp32 tile of  A @ B^T
// Straight-line body (no register double-buffer): round-1 codegen proved this
// lowers to clause'd global_load_b128 + v_wmma with no scratch spill; memory
// latency is hidden by wave oversubscription, not register prefetch.
// ---------------------------------------------------------------------------
__device__ __forceinline__ void wave_gemm_16x64(
    const __bf16* __restrict__ A, int lda,
    const __bf16* __restrict__ Bt, int ldb,
    int K, v8f c[4], int lane)
{
  const int mr = lane & 15;
  const int hf = lane >> 4;
  for (int k0 = 0; k0 < K; k0 += 32) {
    Frag16 a;
    a.h[0] = *(const bf16x8*)(A + mr * lda + k0 + hf * 8);
    a.h[1] = *(const bf16x8*)(A + mr * lda + k0 + 16 + hf * 8);
#pragma unroll
    for (int nt = 0; nt < 4; ++nt) {
      Frag16 b;
      const __bf16* bp = Bt + (nt * 16 + mr) * ldb + k0 + hf * 8;
      b.h[0] = *(const bf16x8*)(bp);
      b.h[1] = *(const bf16x8*)(bp + 16);
      c[nt] = __builtin_amdgcn_wmma_f32_16x16x32_bf16(
          false, a.v, false, b.v, (short)0, c[nt], false, false);
    }
  }
}

// ---------------------------------------------------------------------------
// QKV projection: out = x @ W{q,k,v}^T + b
//   Q, K stored [B,H,S,Dh] bf16 ; V stored transposed [B,H,Dh,S] bf16
// grid = (4096/64, 768/64, 3), block = 128 (4 waves, each 16 rows)
// ---------------------------------------------------------------------------
__global__ __launch_bounds__(128)
__attribute__((amdgpu_waves_per_eu(1, 4)))
void qkv_proj_kernel(const __bf16* __restrict__ xb,
                     const __bf16* __restrict__ Wqb,
                     const __bf16* __restrict__ Wkb,
                     const __bf16* __restrict__ Wvb,
                     const float* __restrict__ bq,
                     const float* __restrict__ bk,
                     const float* __restrict__ bv,
                     __bf16* __restrict__ Q,
                     __bf16* __restrict__ Km,
                     __bf16* __restrict__ Vt)
{
  const int lane = threadIdx.x & 31;
  const int wave = threadIdx.x >> 5;
  const int mr = lane & 15;
  const int hf = lane >> 4;
  const int m0 = blockIdx.x * 64 + wave * 16;
  const int n0 = blockIdx.y * 64;
  const int which = blockIdx.z;                 // 0=Q 1=K 2=V

  const __bf16* W = (which == 0) ? Wqb : (which == 1) ? Wkb : Wvb;
  const float* bias = (which == 0) ? bq : (which == 1) ? bk : bv;

  v8f c[4] = {};
  wave_gemm_16x64(xb + (size_t)m0 * D_MODEL, D_MODEL,
                  W + (size_t)n0 * D_MODEL, D_MODEL,
                  D_MODEL, c, lane);

  const int b  = m0 >> 11;
  const int s0 = m0 & 2047;
#pragma unroll
  for (int nt = 0; nt < 4; ++nt) {
    const int n = n0 + nt * 16 + mr;
    const float bb = bias[n];
    const int h = n >> 6, d = n & 63;
#pragma unroll
    for (int r = 0; r < 8; ++r) {
      const int s = s0 + r + hf * 8;
      const __bf16 vb = f2bf(c[nt][r] + bb);
      if (which == 2) {
        Vt[(((size_t)(b * NHEADS + h) * DHEAD + d) * S_LEN) + s] = vb;
      } else {
        __bf16* dst = (which == 0) ? Q : Km;
        dst[(((size_t)(b * NHEADS + h) * S_LEN + s) * DHEAD) + d] = vb;
      }
    }
  }
}

// ---------------------------------------------------------------------------
// Flash attention per (b, h). Block = 128 threads (4 waves) = 64 q-rows.
// K/V tiles (32 keys) are staged into LDS with GLOBAL_LOAD_ASYNC_TO_LDS_B128,
// double buffered, so all 4 waves share one copy and the async DMA for tile
// i+1 overlaps the WMMAs of tile i.  P is re-laid-out C->A through LDS.
// grid = (S/64, H, B)
// ---------------------------------------------------------------------------
__global__ __launch_bounds__(128)
__attribute__((amdgpu_waves_per_eu(1, 4)))
void attn_kernel(const __bf16* __restrict__ Q,
                 const __bf16* __restrict__ Km,
                 const __bf16* __restrict__ Vt,
                 const int* __restrict__ mask,
                 __bf16* __restrict__ ctxb)
{
  __shared__ __bf16 Ktile[2][32][64];   // [buf][k-local][d]      4KB/buf
  __shared__ __bf16 Vtile[2][64][32];   // [buf][d][k-local]      4KB/buf
  __shared__ __bf16 Pst[4][16][40];     // per-wave P staging

  const int tid  = threadIdx.x;
  const int lane = tid & 31;
  const int wave = tid >> 5;
  const int mr = lane & 15;
  const int hf = lane >> 4;
  const int h = blockIdx.y;
  const int b = blockIdx.z;
  const int q0 = blockIdx.x * 64 + wave * 16;

  const __bf16* Qh = Q  + (size_t)(b * NHEADS + h) * S_LEN * DHEAD;
  const __bf16* Kh = Km + (size_t)(b * NHEADS + h) * S_LEN * DHEAD;
  const __bf16* Vh = Vt + (size_t)(b * NHEADS + h) * DHEAD * S_LEN;

  const uint32_t ldsK0 = lds_addr(&Ktile[0][0][0]);
  const uint32_t ldsV0 = lds_addr(&Vtile[0][0][0]);

  // Async stage of one 32-key tile (K: 4KB contiguous; V^T: 64 rows x 64B)
  auto stage = [&](int kt, int bf) {
    const uint32_t lk = ldsK0 + (uint32_t)bf * 4096u;
    const __bf16* gk = Kh + (size_t)kt * DHEAD;
#pragma unroll
    for (int i = 0; i < 2; ++i) {
      const int c = tid + i * 128;            // 256 x 16B chunks
      asm volatile("global_load_async_to_lds_b128 %0, %1, off"
                   :: "v"(lk + (uint32_t)c * 16u), "v"(gk + c * 8) : "memory");
    }
    const uint32_t lv = ldsV0 + (uint32_t)bf * 4096u;
#pragma unroll
    for (int i = 0; i < 2; ++i) {
      const int c = tid + i * 128;
      const int d = c >> 2, part = c & 3;
      const __bf16* gv = Vh + (size_t)d * S_LEN + kt + part * 8;
      asm volatile("global_load_async_to_lds_b128 %0, %1, off"
                   :: "v"(lv + (uint32_t)(d * 64 + part * 16)), "v"(gv) : "memory");
    }
  };

  // Q fragments: 16 x 64 as two 16x32 A-fragments, resident for whole loop
  Frag16 qf[2];
#pragma unroll
  for (int kc = 0; kc < 2; ++kc) {
    const __bf16* qp = Qh + (size_t)(q0 + mr) * DHEAD + kc * 32;
    qf[kc].h[0] = *(const bf16x8*)(qp + hf * 8);
    qf[kc].h[1] = *(const bf16x8*)(qp + 16 + hf * 8);
  }

  v8f acc[4] = {};
  float mOld[8], lsum[8];
#pragma unroll
  for (int r = 0; r < 8; ++r) { mOld[r] = -1e30f; lsum[r] = 0.f; }

  stage(0, 0);
  asm volatile("s_wait_asynccnt 0x0" ::: "memory");
  __syncthreads();

  int bf = 0;
  for (int kt = 0; kt < S_LEN; kt += 32) {
    if (kt + 32 < S_LEN) {
      stage(kt + 32, bf ^ 1);                  // overlap DMA with compute
      // keep the next mask tile warm in L2 (global_prefetch_b8)
      __builtin_prefetch(mask + (q0 + mr) * S_LEN + kt + 32, 0, 0);
    }

    // ---- scores: 16x32 as two 16x16 C-fragments (K from LDS) ----
    v8f sc[2] = {};
#pragma unroll
    for (int st = 0; st < 2; ++st) {
      const __bf16* kp = &Ktile[bf][st * 16 + mr][0];
#pragma unroll
      for (int kc = 0; kc < 2; ++kc) {
        Frag16 bfr;
        bfr.h[0] = *(const bf16x8*)(kp + kc * 32 + hf * 8);
        bfr.h[1] = *(const bf16x8*)(kp + kc * 32 + 16 + hf * 8);
        sc[st] = __builtin_amdgcn_wmma_f32_16x16x32_bf16(
            false, qf[kc].v, false, bfr.v, (short)0, sc[st], false, false);
      }
    }

    // ---- scale + mask ----
#pragma unroll
    for (int st = 0; st < 2; ++st)
#pragma unroll
      for (int r = 0; r < 8; ++r) {
        const int qrow = q0 + r + hf * 8;
        const int kcol = kt + st * 16 + mr;
        float v = sc[st][r] * 0.125f;               // 1/sqrt(64)
        if (mask[qrow * S_LEN + kcol] == 0) v = -1e9f;
        sc[st][r] = v;
      }

    // ---- online softmax (row stats replicate over each 16-lane group) ----
    float mNew[8];
#pragma unroll
    for (int r = 0; r < 8; ++r) {
      float mx = fmaxf(sc[0][r], sc[1][r]);
#pragma unroll
      for (int off = 1; off < 16; off <<= 1)
        mx = fmaxf(mx, __shfl_xor(mx, off, 32));
      mNew[r] = fmaxf(mOld[r], mx);
    }
    float alpha[8];
#pragma unroll
    for (int r = 0; r < 8; ++r) {
      const float p0 = __expf(sc[0][r] - mNew[r]);
      const float p1 = __expf(sc[1][r] - mNew[r]);
      sc[0][r] = p0; sc[1][r] = p1;
      float rs = p0 + p1;
#pragma unroll
      for (int off = 1; off < 16; off <<= 1)
        rs += __shfl_xor(rs, off, 32);
      alpha[r] = __expf(mOld[r] - mNew[r]);
      lsum[r] = lsum[r] * alpha[r] + rs;
      mOld[r] = mNew[r];
    }
#pragma unroll
    for (int nt = 0; nt < 4; ++nt)
#pragma unroll
      for (int r = 0; r < 8; ++r) acc[nt][r] *= alpha[r];

    // ---- C-layout P -> LDS -> A-layout fragment (bf16) ----
#pragma unroll
    for (int st = 0; st < 2; ++st)
#pragma unroll
      for (int r = 0; r < 8; ++r)
        Pst[wave][r + hf * 8][st * 16 + mr] = f2bf(sc[st][r]);
    asm volatile("s_wait_dscnt 0" ::: "memory");
    Frag16 pf;
    pf.q[0] = *(const bf16x4*)(&Pst[wave][mr][hf * 8]);
    pf.q[1] = *(const bf16x4*)(&Pst[wave][mr][hf * 8 + 4]);
    pf.q[2] = *(const bf16x4*)(&Pst[wave][mr][16 + hf * 8]);
    pf.q[3] = *(const bf16x4*)(&Pst[wave][mr][16 + hf * 8 + 4]);

    // ---- ctx += P @ V  (V^T tile from LDS) ----
#pragma unroll
    for (int nt = 0; nt < 4; ++nt) {
      const __bf16* vp = &Vtile[bf][nt * 16 + mr][hf * 8];
      Frag16 bv;
      bv.h[0] = *(const bf16x8*)(vp);
      bv.h[1] = *(const bf16x8*)(vp + 16);
      acc[nt] = __builtin_amdgcn_wmma_f32_16x16x32_bf16(
          false, pf.v, false, bv.v, (short)0, acc[nt], false, false);
    }

    // make this wave's DMA visible + all waves done with buffer bf
    asm volatile("s_wait_asynccnt 0x0" ::: "memory");
    __syncthreads();
    bf ^= 1;
  }

  // ---- epilogue: normalize and write ctx as bf16 [B, S, H*Dh] ----
#pragma unroll
  for (int nt = 0; nt < 4; ++nt) {
    const int d = h * DHEAD + nt * 16 + mr;
#pragma unroll
    for (int r = 0; r < 8; ++r) {
      const int s = q0 + r + hf * 8;
      const float inv = 1.0f / fmaxf(lsum[r], 1e-20f);
      ctxb[((size_t)(b * S_LEN + s)) * D_MODEL + d] = f2bf(acc[nt][r] * inv);
    }
  }
}

// ---------------------------------------------------------------------------
// Output projection: out = ctx @ Wo^T + bo  (fp32 output)
// ---------------------------------------------------------------------------
__global__ __launch_bounds__(128)
__attribute__((amdgpu_waves_per_eu(1, 4)))
void out_proj_kernel(const __bf16* __restrict__ ctxb,
                     const __bf16* __restrict__ Wob,
                     const float* __restrict__ bo,
                     float* __restrict__ out)
{
  const int lane = threadIdx.x & 31;
  const int wave = threadIdx.x >> 5;
  const int mr = lane & 15;
  const int hf = lane >> 4;
  const int m0 = blockIdx.x * 64 + wave * 16;
  const int n0 = blockIdx.y * 64;

  v8f c[4] = {};
  wave_gemm_16x64(ctxb + (size_t)m0 * D_MODEL, D_MODEL,
                  Wob + (size_t)n0 * D_MODEL, D_MODEL,
                  D_MODEL, c, lane);

#pragma unroll
  for (int nt = 0; nt < 4; ++nt) {
    const int n = n0 + nt * 16 + mr;
    const float bb = bo[n];
#pragma unroll
    for (int r = 0; r < 8; ++r) {
      const int row = m0 + r + hf * 8;
      out[(size_t)row * D_MODEL + n] = c[nt][r] + bb;
    }
  }
}

// ---------------------------------------------------------------------------
extern "C" void kernel_launch(void* const* d_in, const int* in_sizes, int n_in,
                              void* d_out, int out_size, void* d_ws, size_t ws_size,
                              hipStream_t stream) {
  const float* x    = (const float*)d_in[0];
  const int*   mask = (const int*)d_in[1];
  const float* Wq   = (const float*)d_in[2];
  const float* bq   = (const float*)d_in[3];
  const float* Wk   = (const float*)d_in[4];
  const float* bk   = (const float*)d_in[5];
  const float* Wv   = (const float*)d_in[6];
  const float* bv   = (const float*)d_in[7];
  const float* Wo   = (const float*)d_in[8];
  const float* bo   = (const float*)d_in[9];
  float* out = (float*)d_out;

  const size_t nX = (size_t)2 * S_LEN * D_MODEL;
  const size_t nW = (size_t)D_MODEL * D_MODEL;
  const size_t nH = (size_t)2 * NHEADS * S_LEN * DHEAD;

  __bf16* w = (__bf16*)d_ws;
  __bf16* xb  = w; w += nX;
  __bf16* Wqb = w; w += nW;
  __bf16* Wkb = w; w += nW;
  __bf16* Wvb = w; w += nW;
  __bf16* Wob = w; w += nW;
  __bf16* Qb  = w; w += nH;
  __bf16* Kb  = w; w += nH;
  __bf16* Vtb = w; w += nH;
  __bf16* Cb  = w; w += nX;

  auto cvt = [&](const float* s, __bf16* d, size_t n) {
    cvt_bf16_kernel<<<(unsigned)((n / 4 + 255) / 256), 256, 0, stream>>>(s, d, (int)n);
  };
  cvt(x,  xb,  nX);
  cvt(Wq, Wqb, nW);
  cvt(Wk, Wkb, nW);
  cvt(Wv, Wvb, nW);
  cvt(Wo, Wob, nW);

  qkv_proj_kernel<<<dim3(4096 / 64, D_MODEL / 64, 3), 128, 0, stream>>>(
      xb, Wqb, Wkb, Wvb, bq, bk, bv, Qb, Kb, Vtb);

  attn_kernel<<<dim3(S_LEN / 64, NHEADS, 2), 128, 0, stream>>>(
      Qb, Kb, Vtb, mask, Cb);

  out_proj_kernel<<<dim3(4096 / 64, D_MODEL / 64, 1), 128, 0, stream>>>(
      Cb, Wob, bo, out);
}
